// ModulatedConv_4483945857634
// MI455X (gfx1250) — matrix operand order repaired and checked by reading, not compiled, not run on hardware
//
#include <hip/hip_runtime.h>
#include <hip/hip_bf16.h>

// ---------------------------------------------------------------------------
// Modulated transposed conv (StyleGAN2 style) for MI455X / gfx1250.
// 4 output-parity quadrants, each an implicit GEMM on v_wmma_f32_16x16x32_bf16.
// A-tiles (weights) staged into LDS by the Tensor Data Mover (tensor_load_to_lds,
// TENSORcnt, double-buffered, TDM LDS-padding for conflict-free ds_load_b128);
// B (modulated x) gathered per-lane from global; demod applied in epilogue.
// ---------------------------------------------------------------------------

typedef __attribute__((ext_vector_type(16))) __bf16 v16bf;
typedef __attribute__((ext_vector_type(8)))  float  v8f;
typedef unsigned int v4u __attribute__((ext_vector_type(4)));
typedef int          v8i __attribute__((ext_vector_type(8)));
typedef int          v4i __attribute__((ext_vector_type(4)));

#define IC    512
#define OC    512
#define BATCH 16
#define HH    32
#define NPIX  1024   // 32*32 input pixels
#define HO    65     // output spatial size

// A-tile in LDS: 128 rows x 32 bf16, TDM pads +16B per 64B row -> 80B pitch
#define TILE_PITCH 40          // ushorts per row (80 bytes)

__device__ __forceinline__ unsigned short f2bf(float f) {
  unsigned int u = __float_as_uint(f);
  unsigned int r = (u + 0x7FFFu + ((u >> 16) & 1u)) >> 16;
  return (unsigned short)r;
}

union Frag {
  v16bf v;
  uint4 u[2];
};

// ---------------------------------------------------------------------------
// K1: style vector  s[b,ic] = w[b,:] @ (lin_w[ic,:] * sqrt(2/512)) + lin_b[ic]
// ---------------------------------------------------------------------------
__global__ void k_style(const float* __restrict__ w, const float* __restrict__ lin_w,
                        const float* __restrict__ lin_b, float* __restrict__ s) {
  int gid = blockIdx.x * blockDim.x + threadIdx.x;   // 8192
  int b = gid >> 9, ic = gid & 511;
  float acc = lin_b[ic];
  const float* wr = w + b * 512;
  const float* lr = lin_w + ic * 512;
  const float c1 = 0.0625f;                          // sqrt(2/512)
  for (int j = 0; j < 512; ++j) acc += wr[j] * (lr[j] * c1);
  s[gid] = acc;
}

// ---------------------------------------------------------------------------
// K2: wtb[t][oc][ic] = bf16(weight * sqrt(2/4608)),  q[oc][ic] = sum_t wt^2
// ---------------------------------------------------------------------------
__global__ void k_weight(const float* __restrict__ weight,
                         unsigned short* __restrict__ wtb, float* __restrict__ q) {
  int gid = blockIdx.x * blockDim.x + threadIdx.x;   // 262144 ; gid = oc*512+ic
  const float c2 = 1.0f / 48.0f;                     // sqrt(2/4608)
  float qv = 0.f;
#pragma unroll
  for (int t = 0; t < 9; ++t) {
    float v = weight[(size_t)gid * 9 + t] * c2;
    qv += v * v;
    wtb[(size_t)t * (OC * IC) + gid] = f2bf(v);
  }
  q[gid] = qv;
}

// ---------------------------------------------------------------------------
// K3: demod  d[b,oc] = rsqrt( sum_ic s[b,ic]^2 * q[oc,ic] + 1e-8 )
// ---------------------------------------------------------------------------
__global__ void k_demod(const float* __restrict__ s, const float* __restrict__ q,
                        float* __restrict__ d) {
  int gid = blockIdx.x * blockDim.x + threadIdx.x;   // 8192
  int b = gid >> 9, oc = gid & 511;
  float acc = 1e-8f;
  const float* sr = s + b * 512;
  const float* qr = q + (size_t)oc * 512;
  for (int ic = 0; ic < 512; ++ic) {
    float sv = sr[ic];
    acc += sv * sv * qr[ic];
  }
  d[gid] = rsqrtf(acc);
}

// ---------------------------------------------------------------------------
// K4: modulate + transpose: xT[b][pix][ic] = bf16( s[b,ic] * x[b,ic,pix] )
// ---------------------------------------------------------------------------
__global__ void k_xmod(const float* __restrict__ x, const float* __restrict__ s,
                       unsigned short* __restrict__ xT) {
  __shared__ float tile[32][33];
  const int b = blockIdx.z, ic0 = blockIdx.y * 32, p0 = blockIdx.x * 32;
  const int col = threadIdx.x & 31, row0 = threadIdx.x >> 5;
#pragma unroll
  for (int r = 0; r < 4; ++r) {
    int icl = row0 + r * 8;
    tile[icl][col] = x[((size_t)(b * IC + ic0 + icl)) * NPIX + p0 + col] *
                     s[b * IC + ic0 + icl];
  }
  __syncthreads();
#pragma unroll
  for (int r = 0; r < 4; ++r) {
    int pl = row0 + r * 8;
    xT[((size_t)(b * NPIX + p0 + pl)) * IC + ic0 + col] = f2bf(tile[col][pl]);
  }
}

// ---------------------------------------------------------------------------
// TDM issue: 2D tile (32 cols x 128 rows) of bf16, row stride 512 elements,
// destination LDS with +16B pad per 64B row (bank-conflict-free 80B pitch).
// ---------------------------------------------------------------------------
#if __has_builtin(__builtin_amdgcn_tensor_load_to_lds)
#define HAVE_TDM 1
__device__ __forceinline__ void tdm_load_tile(unsigned lds_off,
                                              const unsigned short* gptr) {
  unsigned long long ga = (unsigned long long)(uintptr_t)gptr;
  v4u g0;
  g0.x = 1u;                                            // count=1, user desc
  g0.y = lds_off;                                       // lds_addr (bytes)
  g0.z = (unsigned)(ga & 0xFFFFFFFFu);                  // global_addr[31:0]
  g0.w = (unsigned)((ga >> 32) & 0x01FFFFFFu)           // global_addr[56:32]
       | (2u << 30);                                    // type=2 (image)
  v8i g1;
  g1[0] = (1 << 16)        // data_size = 2 bytes
        | (1 << 20)        // pad_enable
        | (3 << 22)        // pad_interval: 16 DWORDs (64B)
        | (3 << 25);       // pad_amount: 4 DWORDs (16B)
  g1[1] = (IC & 0xFFFF) << 16;          // tensor_dim0[15:0] @ bits 63:48
  g1[2] = (OC & 0xFFFF) << 16;          // tensor_dim1[15:0] @ bits 95:80
  g1[3] = 32 << 16;                     // tile_dim0 = 32   @ bits 127:112
  g1[4] = 128;                          // tile_dim1 = 128  @ bits 143:128
  g1[5] = IC;                           // tensor_dim0_stride lo
  g1[6] = 0;
  g1[7] = 0;
  v4i gz = {0, 0, 0, 0};
#if __clang_major__ >= 23
  v8i gz8 = {0, 0, 0, 0, 0, 0, 0, 0};
  __builtin_amdgcn_tensor_load_to_lds(g0, g1, gz, gz, gz8, 0);
#else
  __builtin_amdgcn_tensor_load_to_lds(g0, g1, gz, gz, 0);
#endif
}
#else
#define HAVE_TDM 0
#endif

__device__ __forceinline__ void tap_decode(int tap, int py, int px,
                                           int& ky, int& kx) {
  int tx = px ? 0 : (tap & 1);
  int ty = py ? 0 : (px ? tap : (tap >> 1));
  ky = py ? 1 : ty * 2;
  kx = px ? 1 : tx * 2;
}

// ---------------------------------------------------------------------------
// K5: main implicit-GEMM per parity quadrant.
//   grid = (4 Mtiles * 18 Ntiles, batch, quadrant), block = 256 (8 waves).
//   Block tile M=128 x N=64; wave tile 64x16 => 4 v8f accumulators.
//   A staged in LDS by TDM (double buffered); fragments via ds_load_b128.
// ---------------------------------------------------------------------------
__global__ __launch_bounds__(256) void k_conv_wmma(
    const unsigned short* __restrict__ xT, const unsigned short* __restrict__ wtb,
    const float* __restrict__ dcoef, float* __restrict__ out) {
  __shared__ __align__(16) unsigned short atile[2][128 * TILE_PITCH];

  const int b = blockIdx.y;
  const int z = blockIdx.z;                 // quadrant
  const int py = 1 - (z >> 1), px = 1 - (z & 1);
  const int Ny = py ? 32 : 33, Nx = px ? 32 : 33;
  const int Nq = Ny * Nx;                   // 1024 / 1056 / 1056 / 1089
  const int mt = blockIdx.x & 3;            // 4 M-tiles of 128
  const int ntile = blockIdx.x >> 2;        // up to 18 N-tiles of 64
  if (ntile * 64 >= Nq) return;             // block-uniform early exit

  const int wave = threadIdx.x >> 5;
  const int lane = threadIdx.x & 31;
  const int wm = wave >> 2;                 // 0..1 : 64-row M half
  const int wn = wave & 3;                  // 0..3 : 16-col N quarter
  const int nlo = lane & 15;
  const int khalf = lane >> 4;

  const int mbase = mt * 128 + wm * 64;
  const int nbase = ntile * 64 + wn * 16;
  const int n = nbase + nlo;
  const bool nval = (n < Nq);
  const int ny = nval ? (n / Nx) : 0;
  const int nx = nval ? (n % Nx) : 0;

  const v8f vzero = {0.f, 0.f, 0.f, 0.f, 0.f, 0.f, 0.f, 0.f};
  v8f acc[4];
#pragma unroll
  for (int i = 0; i < 4; ++i) acc[i] = vzero;

  const int ntaps = (py ? 1 : 2) * (px ? 1 : 2);
  const int nsteps = ntaps << 4;            // ntaps * (512/32)

  for (int s = 0; s < nsteps; ++s) {
    const int tap = s >> 4;
    const int kc = (s & 15) << 5;
    int ky, kx;
    tap_decode(tap, py, px, ky, kx);

    __syncthreads();   // readers of buffer (s&1) from step s-2 are done
#if HAVE_TDM
    if (wave == 0) {
      if (s == 0) {
        tdm_load_tile((unsigned)(uintptr_t)&atile[0][0],
                      wtb + ((size_t)((ky * 3 + kx) * OC + mt * 128)) * IC + kc);
      }
      if (s + 1 < nsteps) {
        const int s2 = s + 1;
        int ky2, kx2;
        tap_decode(s2 >> 4, py, px, ky2, kx2);
        tdm_load_tile((unsigned)(uintptr_t)&atile[s2 & 1][0],
                      wtb + ((size_t)((ky2 * 3 + kx2) * OC + mt * 128)) * IC +
                          ((s2 & 15) << 5));
        __builtin_amdgcn_s_wait_tensorcnt(1);   // current buffer complete
      } else {
        __builtin_amdgcn_s_wait_tensorcnt(0);
      }
    }
#else
    {   // fallback staging: 256 threads copy 128x32 bf16 tile into padded LDS
      const int r = threadIdx.x >> 1, h = threadIdx.x & 1;
      const unsigned short* src =
          wtb + ((size_t)((ky * 3 + kx) * OC + mt * 128 + r)) * IC + kc + h * 16;
      uint4 a0 = *reinterpret_cast<const uint4*>(src);
      uint4 a1 = *reinterpret_cast<const uint4*>(src + 8);
      *reinterpret_cast<uint4*>(&atile[s & 1][r * TILE_PITCH + h * 16]) = a0;
      *reinterpret_cast<uint4*>(&atile[s & 1][r * TILE_PITCH + h * 16 + 8]) = a1;
    }
#endif
    __syncthreads();   // buffer (s&1) ready

    // ---- B fragment: per-lane gather of one modulated-x column slice ----
    const int iy = ny - (ky == 2);
    const int ix = nx - (kx == 2);
    const bool v = nval && (iy >= 0) && (iy < HH) && (ix >= 0) && (ix < HH);
    const int pix = v ? (iy * HH + ix) : 0;
    const unsigned short* xcol =
        xT + ((size_t)(b * NPIX + pix)) * IC + khalf * 16 + kc;
    Frag xb;
    if (v) {
      xb.u[0] = *reinterpret_cast<const uint4*>(xcol);
      xb.u[1] = *reinterpret_cast<const uint4*>(xcol + 8);
      if (kc + 32 < IC) __builtin_prefetch(xcol + 32, 0, 1);
    } else {
      xb.u[0] = make_uint4(0u, 0u, 0u, 0u);
      xb.u[1] = make_uint4(0u, 0u, 0u, 0u);
    }

    // ---- A fragments from LDS (80B pitch => conflict-free b128 reads) ----
    const unsigned short* lb =
        &atile[s & 1][(wm * 64 + nlo) * TILE_PITCH + khalf * 16];
#pragma unroll
    for (int mf = 0; mf < 4; ++mf) {
      const unsigned short* ap = lb + mf * 16 * TILE_PITCH;
      Frag af;
      af.u[0] = *reinterpret_cast<const uint4*>(ap);
      af.u[1] = *reinterpret_cast<const uint4*>(ap + 8);
      acc[mf] = __builtin_amdgcn_wmma_f32_16x16x32_bf16(
          false, af.v, false, xb.v, (short)0, acc[mf], false, false);
    }
  }

  if (!nval) return;  // per-lane exit after all WMMAs / barriers
  const int oy = 2 * ny + py;
  const int ox = 2 * nx + px;
#pragma unroll
  for (int mf = 0; mf < 4; ++mf) {
#pragma unroll
    for (int e = 0; e < 8; ++e) {
      const int oc = mbase + mf * 16 + khalf * 8 + e;
      out[(((size_t)(b * OC + oc)) * HO + oy) * HO + ox] =
          dcoef[b * OC + oc] * acc[mf][e];
    }
  }
}

// ---------------------------------------------------------------------------
// Host launcher
// ---------------------------------------------------------------------------
extern "C" void kernel_launch(void* const* d_in, const int* in_sizes, int n_in,
                              void* d_out, int out_size, void* d_ws, size_t ws_size,
                              hipStream_t stream) {
  (void)in_sizes; (void)n_in; (void)out_size; (void)ws_size;
  const float* x      = (const float*)d_in[0];  // (16,512,32,32)
  const float* w      = (const float*)d_in[1];  // (16,512)
  const float* weight = (const float*)d_in[2];  // (1,512,512,3,3)
  const float* lin_w  = (const float*)d_in[3];  // (512,512)
  const float* lin_b  = (const float*)d_in[4];  // (512,)
  float* out = (float*)d_out;                   // (16,512,65,65)

  char* ws = (char*)d_ws;
  float* s    = (float*)(ws + 0);               //  32 KB : s[b][ic]
  float* dmod = (float*)(ws + 32768);           //  32 KB : d[b][oc]
  float* q    = (float*)(ws + 65536);           //   1 MB : q[oc][ic]
  unsigned short* wtb = (unsigned short*)(ws + 1114112);   // 4.5 MB bf16 [9][oc][ic]
  unsigned short* xT  = (unsigned short*)(ws + 5832704);   //  16 MB bf16 [b][pix][ic]

  k_style<<<32, 256, 0, stream>>>(w, lin_w, lin_b, s);
  k_weight<<<1024, 256, 0, stream>>>(weight, wtb, q);
  k_demod<<<32, 256, 0, stream>>>(s, q, dmod);
  k_xmod<<<dim3(32, 16, 16), 256, 0, stream>>>(x, s, xT);
  k_conv_wmma<<<dim3(4 * 18, BATCH, 4), 256, 0, stream>>>(xT, wtb, dmod, out);
}
